// LrFeatureUpScaler_77618648973641
// MI455X (gfx1250) — compile-verified, matrix operand order checked by reading, not run on hardware
//
#include <hip/hip_runtime.h>
#include <cstdint>

#define NN 4096
#define HH 8
#define CC 512
#define DD 64

typedef __attribute__((ext_vector_type(16))) __bf16 v16bf;
typedef __attribute__((ext_vector_type(8)))  __bf16 v8bf;
typedef __attribute__((ext_vector_type(8)))  float  v8f;

__device__ __forceinline__ unsigned short f32_bf16(float f) {
  union { float f; unsigned u; } v; v.f = f;
  unsigned r = v.u + 0x7fffu + ((v.u >> 16) & 1u);   // round-to-nearest-even
  return (unsigned short)(r >> 16);
}
__device__ __forceinline__ float bf16_f32(unsigned short u) {
  union { unsigned u; float f; } v; v.u = ((unsigned)u) << 16;
  return v.f;
}
__device__ __forceinline__ v8f wmma_bf16(v16bf a, v16bf b, v8f c) {
  // D = A(16x32 bf16) x B(32x16 bf16) + C(16x16 f32)
  return __builtin_amdgcn_wmma_f32_16x16x32_bf16(false, a, false, b, (short)0, c, false, false);
}
// 16 contiguous bf16 from LDS (16B-aligned) -> A/B fragment
__device__ __forceinline__ v16bf lds_frag16(const unsigned short* p) {
  const v8bf* q = (const v8bf*)p;
  v8bf lo = q[0], hi = q[1];
  return __builtin_shufflevector(lo, hi, 0,1,2,3,4,5,6,7,8,9,10,11,12,13,14,15);
}

// ---- CDNA5 async memory->LDS copy (ASYNCcnt), bypasses VGPRs ----
// LDS destination = low 32 bits of the generic pointer (aperture maps addr[31:0] to LDS).
__device__ __forceinline__ void async_load_b128(void* lds_ptr, const void* gptr) {
  unsigned lds_off = (unsigned)(uintptr_t)lds_ptr;
  unsigned long long ga = (unsigned long long)(uintptr_t)gptr;
  asm volatile("global_load_async_to_lds_b128 %0, %1, off"
               :: "v"(lds_off), "v"(ga)
               : "memory");
}
__device__ __forceinline__ void wait_async0() {
  asm volatile("s_wait_asynccnt 0" ::: "memory");
}

// ---- DPP16 16-lane butterfly reductions (no LDS traffic, pure VALU) ----
template <int CTRL>
__device__ __forceinline__ float dpp_f(float x) {
  int i = __float_as_int(x);
  int r = __builtin_amdgcn_update_dpp(i, i, CTRL, 0xF, 0xF, true);
  return __int_as_float(r);
}
__device__ __forceinline__ float rowmax16(float x) {
  x = fmaxf(x, dpp_f<0xB1>(x));    // quad_perm [1,0,3,2]  (xor 1)
  x = fmaxf(x, dpp_f<0x4E>(x));    // quad_perm [2,3,0,1]  (xor 2)
  x = fmaxf(x, dpp_f<0x124>(x));   // row_ror:4
  x = fmaxf(x, dpp_f<0x128>(x));   // row_ror:8
  return x;
}
__device__ __forceinline__ float rowsum16(float x) {
  x += dpp_f<0xB1>(x);
  x += dpp_f<0x4E>(x);
  x += dpp_f<0x124>(x);
  x += dpp_f<0x128>(x);
  return x;
}

// ---------------- packing kernels ----------------
__global__ __launch_bounds__(256) void cvt_x_kernel(const float* __restrict__ x,
                                                    unsigned short* __restrict__ xb) {
  size_t idx = (size_t)blockIdx.x * 256 + threadIdx.x;   // grid covers NN*NN exactly
  xb[idx] = f32_bf16(x[idx]);
}

__global__ __launch_bounds__(256) void pack_w_kernel(const float* __restrict__ Wq, const float* __restrict__ Wk,
                                                     const float* __restrict__ Wv, const float* __restrict__ Ws,
                                                     unsigned short* __restrict__ Tq, unsigned short* __restrict__ Tk,
                                                     unsigned short* __restrict__ Tv, unsigned short* __restrict__ Ts) {
  size_t idx = (size_t)blockIdx.x * 256 + threadIdx.x;   // over NN*CC (c fastest over n)
  int n = (int)(idx / CC), c = (int)(idx % CC);
  size_t o = (size_t)c * NN + n;                         // transposed: T[c][n] = W[n][c]
  Tq[o] = f32_bf16(Wq[idx]); Tk[o] = f32_bf16(Wk[idx]);
  Tv[o] = f32_bf16(Wv[idx]); Ts[o] = f32_bf16(Ws[idx]);
}

// ---------------- projection GEMM: out = x @ W + b ----------------
// MODE: 0->Qb bf16 row-major, 1->Kb bf16 row-major, 2->Vt bf16 transposed [c][n], 3->skip f32 row-major
template <int MODE>
__global__ __launch_bounds__(256) void proj_kernel(const unsigned short* __restrict__ xb,
                                                   const unsigned short* __restrict__ Wt,
                                                   const float* __restrict__ bias,
                                                   void* __restrict__ out) {
  __shared__ __align__(16) unsigned short As[128 * 40];  // 128 rows x 32 k (+8 pad)
  __shared__ __align__(16) unsigned short Bs[64 * 40];   // 64 cols(c) x 32 k (+8 pad)

  const int bm = blockIdx.x & 31;        // 4096/128 = 32
  const int bn = blockIdx.x >> 5;        // 512/64  = 8
  const int w  = threadIdx.x >> 5;
  const int l  = threadIdx.x & 31;
  const int half = l >> 4, lm = l & 15;
  const int wm = w >> 1, wn = w & 1;     // 4x2 wave grid, each wave 32x32
  const int rowBase = bm * 128, cBase = bn * 64;

  v8f zero = {};
  v8f acc[2][2];
  acc[0][0] = zero; acc[0][1] = zero; acc[1][0] = zero; acc[1][1] = zero;

  const int t  = threadIdx.x;
  const int r  = t >> 1, kc = (t & 1) * 16;   // A tile: 128 rows x 32 k
  const int rb = t >> 2, kb2 = (t & 3) * 8;   // B tile: 64 rows x 32 k

  for (int k0 = 0; k0 < NN; k0 += 32) {
    __syncthreads();
    async_load_b128(As + r * 40 + kc,     xb + (size_t)(rowBase + r) * NN + k0 + kc);
    async_load_b128(As + r * 40 + kc + 8, xb + (size_t)(rowBase + r) * NN + k0 + kc + 8);
    async_load_b128(Bs + rb * 40 + kb2,   Wt + (size_t)(cBase + rb) * NN + k0 + kb2);
    wait_async0();
    __syncthreads();
    v16bf af[2], bf2[2];
#pragma unroll
    for (int sm = 0; sm < 2; sm++)
      af[sm] = lds_frag16(As + (wm * 32 + sm * 16 + lm) * 40 + half * 16);
#pragma unroll
    for (int sn = 0; sn < 2; sn++)
      bf2[sn] = lds_frag16(Bs + (wn * 32 + sn * 16 + lm) * 40 + half * 16);
#pragma unroll
    for (int sm = 0; sm < 2; sm++)
#pragma unroll
      for (int sn = 0; sn < 2; sn++)
        acc[sm][sn] = wmma_bf16(af[sm], bf2[sn], acc[sm][sn]);
  }

#pragma unroll
  for (int sm = 0; sm < 2; sm++)
#pragma unroll
    for (int sn = 0; sn < 2; sn++) {
      const int c = cBase + wn * 32 + sn * 16 + lm;
      const float bc = bias[c];
#pragma unroll
      for (int g = 0; g < 8; g++) {
        const int i = rowBase + wm * 32 + sm * 16 + g + 8 * half;
        float v = acc[sm][sn][g] + bc;
        if (MODE == 3)      ((float*)out)[(size_t)i * CC + c] = v;
        else if (MODE == 2) ((unsigned short*)out)[(size_t)c * NN + i] = f32_bf16(v);
        else                ((unsigned short*)out)[(size_t)i * CC + c] = f32_bf16(v);
      }
    }
}

// ---------------- qe[i,h] = sum_d q[i,h,d] * we[h,d] ----------------
__global__ __launch_bounds__(256) void qe_kernel(const unsigned short* __restrict__ Qb,
                                                 const float* __restrict__ we,
                                                 float* __restrict__ qe) {
  int idx = blockIdx.x * 256 + threadIdx.x;  // NN*HH
  int i = idx >> 3, h = idx & 7;
  float s = 0.f;
#pragma unroll 8
  for (int d = 0; d < DD; d++)
    s += bf16_f32(Qb[(size_t)i * CC + h * DD + d]) * we[h * DD + d];
  qe[idx] = s;
}

// ---------------- flash attention (one head x 128 rows per block) ----------------
__global__ __launch_bounds__(256) void attn_kernel(const float* __restrict__ x,
                                                   const unsigned short* __restrict__ Qb,
                                                   const unsigned short* __restrict__ Kb,
                                                   const unsigned short* __restrict__ Vt,
                                                   const float* __restrict__ qe,
                                                   const float* __restrict__ we,
                                                   const float* __restrict__ sk,
                                                   float* __restrict__ y) {
  __shared__ __align__(16) unsigned short Ks[32 * 72];      // [j_local][d] +8 pad
  __shared__ __align__(16) unsigned short Vs[64 * 40];      // [d_local][j] +8 pad (transposed V)
  __shared__ __align__(16) unsigned short Ps[8 * 16 * 40];  // per-wave P tile [i 16][j 32] +8 pad

  const int h  = blockIdx.x % HH;
  const int ib = blockIdx.x / HH;        // 0..31
  const int w  = threadIdx.x >> 5;
  const int l  = threadIdx.x & 31;
  const int half = l >> 4, lm = l & 15;
  const int iBase = ib * 128 + w * 16;

  // Q fragments for this wave's 16 rows (A layout: lane=row, K split by lane half)
  v16bf qa0 = *(const v16bf*)(Qb + (size_t)(iBase + lm) * CC + h * DD + half * 16);
  v16bf qa1 = *(const v16bf*)(Qb + (size_t)(iBase + lm) * CC + h * DD + 32 + half * 16);

  float qer[8];
#pragma unroll
  for (int g = 0; g < 8; g++) qer[g] = qe[(size_t)(iBase + half * 8 + g) * HH + h];
  float wer[4];
#pragma unroll
  for (int t = 0; t < 4; t++) wer[t] = we[h * DD + t * 16 + lm];

  v8f zero = {};
  v8f O[4]; O[0] = zero; O[1] = zero; O[2] = zero; O[3] = zero;
  float m[8], lsum[8], ps[8];
#pragma unroll
  for (int g = 0; g < 8; g++) { m[g] = -1e30f; lsum[g] = 0.f; ps[g] = 0.f; }

  const int tt = threadIdx.x;
  const int jl = tt >> 3, dc = (tt & 7) * 8;   // K tile: 32 j x 64 d
  const int dl = tt >> 2, jc = (tt & 3) * 8;   // V tile: 64 d x 32 j

  for (int j0 = 0; j0 < NN; j0 += 32) {
    __syncthreads();
    async_load_b128(Ks + jl * 72 + dc, Kb + (size_t)(j0 + jl) * CC + h * DD + dc);
    async_load_b128(Vs + dl * 40 + jc, Vt + (size_t)(h * DD + dl) * NN + j0 + jc);
    wait_async0();
    __syncthreads();

    // S = Q K^T  (two 16x16 j-tiles, K-dim d=64 in two chunks)
    v8f S0 = zero, S1 = zero;
    {
      v16bf kb;
      kb = lds_frag16(Ks + lm * 72 + half * 16);             S0 = wmma_bf16(qa0, kb, S0);
      kb = lds_frag16(Ks + lm * 72 + 32 + half * 16);        S0 = wmma_bf16(qa1, kb, S0);
      kb = lds_frag16(Ks + (16 + lm) * 72 + half * 16);      S1 = wmma_bf16(qa0, kb, S1);
      kb = lds_frag16(Ks + (16 + lm) * 72 + 32 + half * 16); S1 = wmma_bf16(qa1, kb, S1);
    }

    // edge term xa(i,j)=x[j,i], scale 1/sqrt(64), online softmax (DPP row reductions)
    float xa0[8], xa1[8], a[8];
#pragma unroll
    for (int g = 0; g < 8; g++) {
      int ig = iBase + half * 8 + g;
      xa0[g] = x[(size_t)(j0 + lm) * NN + ig];
      xa1[g] = x[(size_t)(j0 + 16 + lm) * NN + ig];
    }
#pragma unroll
    for (int g = 0; g < 8; g++) {
      float s0 = (S0[g] + xa0[g] * qer[g]) * 0.125f;
      float s1 = (S1[g] + xa1[g] * qer[g]) * 0.125f;
      S0[g] = s0; S1[g] = s1;
      float mc = rowmax16(fmaxf(s0, s1));
      float mn = fmaxf(m[g], mc);
      a[g] = __expf(m[g] - mn);
      m[g] = mn;
    }
    unsigned short* pw = Ps + w * (16 * 40);
#pragma unroll
    for (int g = 0; g < 8; g++) {
      float p0 = __expf(S0[g] - m[g]);
      float p1 = __expf(S1[g] - m[g]);
      lsum[g] = lsum[g] * a[g] + p0 + p1;
      ps[g]   = ps[g] * a[g] + p0 * xa0[g] + p1 * xa1[g];
#pragma unroll
      for (int t = 0; t < 4; t++) O[t][g] *= a[g];
      int r = g + 8 * half;
      pw[r * 40 + lm]      = f32_bf16(p0);
      pw[r * 40 + 16 + lm] = f32_bf16(p1);
    }
    __syncthreads();   // make P tile visible in A-fragment layout

    v16bf pf = lds_frag16(pw + lm * 40 + half * 16);
#pragma unroll
    for (int t = 0; t < 4; t++) {
      v16bf vb = lds_frag16(Vs + (t * 16 + lm) * 40 + half * 16);
      O[t] = wmma_bf16(pf, vb, O[t]);
    }
  }

  // finalize: normalize by l, add s*we + skip, write pre-norm y
#pragma unroll
  for (int g = 0; g < 8; g++) {
    float ls = rowsum16(lsum[g]);
    float pp = rowsum16(ps[g]);
    float inv = 1.0f / ls;
    float sv = pp * inv;
    int ig = iBase + half * 8 + g;
#pragma unroll
    for (int t = 0; t < 4; t++) {
      int c = h * DD + t * 16 + lm;
      y[(size_t)ig * CC + c] = O[t][g] * inv + sv * wer[t] + sk[(size_t)ig * CC + c];
    }
  }
}

// ---------------- GraphNorm stats: S1=sum(y), S2=sum(y^2) per channel ----------------
__global__ __launch_bounds__(256) void stats_kernel(const float* __restrict__ y,
                                                    float* __restrict__ S1, float* __restrict__ S2) {
  int b = blockIdx.x;           // 64 blocks x 64 rows
  int t = threadIdx.x;
#pragma unroll
  for (int cc = 0; cc < 2; cc++) {
    int c = t + cc * 256;
    float s1 = 0.f, s2 = 0.f;
    for (int r = 0; r < 64; r++) {
      float v = y[(size_t)(b * 64 + r) * CC + c];
      s1 += v; s2 += v * v;
    }
    atomicAdd(&S1[c], s1);
    atomicAdd(&S2[c], s2);
  }
}

__global__ void gn_kernel(const float* __restrict__ S1, const float* __restrict__ S2,
                          const float* __restrict__ gw, const float* __restrict__ gb,
                          const float* __restrict__ gms,
                          float* __restrict__ scaleb, float* __restrict__ shiftb) {
  int c = blockIdx.x * 256 + threadIdx.x;
  if (c >= CC) return;
  float mean = S1[c] * (1.0f / NN);
  float aa = gms[c] * mean;
  float var = S2[c] * (1.0f / NN) - 2.f * aa * mean + aa * aa;
  float sc = gw[c] * rsqrtf(var + 1e-5f);
  scaleb[c] = sc;
  shiftb[c] = gb[c] - aa * sc;
}

// ---------------- per-row affine + L2 normalize (in place on y) ----------------
__global__ __launch_bounds__(256) void l2_kernel(float* __restrict__ y,
                                                 const float* __restrict__ scaleb,
                                                 const float* __restrict__ shiftb) {
  __shared__ float red[8];
  int i = blockIdx.x, t = threadIdx.x;
  float z0 = y[(size_t)i * CC + t]       * scaleb[t]       + shiftb[t];
  float z1 = y[(size_t)i * CC + t + 256] * scaleb[t + 256] + shiftb[t + 256];
  float s = z0 * z0 + z1 * z1;
  s = rowsum16(s);                 // 16-lane DPP sum
  s += __shfl_xor(s, 16, 32);      // cross-half
  if ((t & 31) == 0) red[t >> 5] = s;
  __syncthreads();
  if (t < 8) {
    float v = red[t];
    v += __shfl_xor(v, 1, 32); v += __shfl_xor(v, 2, 32); v += __shfl_xor(v, 4, 32);
    if (t == 0) red[0] = v;
  }
  __syncthreads();
  float inv = rsqrtf(red[0]);
  y[(size_t)i * CC + t]       = z0 * inv;
  y[(size_t)i * CC + t + 256] = z1 * inv;
}

// ---------------- launch ----------------
extern "C" void kernel_launch(void* const* d_in, const int* in_sizes, int n_in,
                              void* d_out, int out_size, void* d_ws, size_t ws_size,
                              hipStream_t stream) {
  (void)in_sizes; (void)n_in; (void)out_size; (void)ws_size;
  const float* x   = (const float*)d_in[0];
  const float* Wq  = (const float*)d_in[1];
  const float* bq  = (const float*)d_in[2];
  const float* Wk  = (const float*)d_in[3];
  const float* bk  = (const float*)d_in[4];
  const float* Wv  = (const float*)d_in[5];
  const float* bv  = (const float*)d_in[6];
  const float* we  = (const float*)d_in[7];
  const float* Wsk = (const float*)d_in[8];
  const float* bsk = (const float*)d_in[9];
  const float* gw  = (const float*)d_in[10];
  const float* gb  = (const float*)d_in[11];
  const float* gms = (const float*)d_in[12];
  float* y = (float*)d_out;

  char* p = (char*)d_ws;
  auto take = [&](size_t bytes) { void* r = (void*)p; p += (bytes + 255) & ~(size_t)255; return r; };
  unsigned short* xb = (unsigned short*)take((size_t)NN * NN * 2);   // 32 MB
  unsigned short* Tq = (unsigned short*)take((size_t)CC * NN * 2);   // 4 MB each
  unsigned short* Tk = (unsigned short*)take((size_t)CC * NN * 2);
  unsigned short* Tv = (unsigned short*)take((size_t)CC * NN * 2);
  unsigned short* Ts = (unsigned short*)take((size_t)CC * NN * 2);
  unsigned short* Qb = (unsigned short*)take((size_t)NN * CC * 2);
  unsigned short* Kb = (unsigned short*)take((size_t)NN * CC * 2);
  unsigned short* Vt = (unsigned short*)take((size_t)NN * CC * 2);
  float* sk     = (float*)take((size_t)NN * CC * 4);                 // 8 MB
  float* qe     = (float*)take((size_t)NN * HH * 4);
  float* S1     = (float*)take(CC * 4);
  float* S2     = (float*)take(CC * 4);
  float* scaleb = (float*)take(CC * 4);
  float* shiftb = (float*)take(CC * 4);

  cvt_x_kernel<<<(NN * (size_t)NN) / 256, 256, 0, stream>>>(x, xb);
  pack_w_kernel<<<(NN * CC) / 256, 256, 0, stream>>>(Wq, Wk, Wv, Wsk, Tq, Tk, Tv, Ts);

  proj_kernel<0><<<256, 256, 0, stream>>>(xb, Tq, bq, (void*)Qb);
  proj_kernel<1><<<256, 256, 0, stream>>>(xb, Tk, bk, (void*)Kb);
  proj_kernel<2><<<256, 256, 0, stream>>>(xb, Tv, bv, (void*)Vt);
  proj_kernel<3><<<256, 256, 0, stream>>>(xb, Ts, bsk, (void*)sk);

  qe_kernel<<<(NN * HH) / 256, 256, 0, stream>>>(Qb, we, qe);
  attn_kernel<<<HH * (NN / 128), 256, 0, stream>>>(x, Qb, Kb, Vt, qe, we, sk, y);

  hipMemsetAsync(S1, 0, CC * 4, stream);
  hipMemsetAsync(S2, 0, CC * 4, stream);
  stats_kernel<<<64, 256, 0, stream>>>(y, S1, S2);
  gn_kernel<<<2, 256, 0, stream>>>(S1, S2, gw, gb, gms, scaleb, shiftb);
  l2_kernel<<<NN, 256, 0, stream>>>(y, scaleb, shiftb);
}
